// DynamicEdgeConvGNN_62474594287978
// MI455X (gfx1250) — compile-verified, matrix operand order checked by reading, not run on hardware
//
#include <hip/hip_runtime.h>

// ---------------------------------------------------------------------------
// DGCNN on MI455X (gfx1250): WMMA-f16 GEMM pipeline, 16x64 register-blocked.
//   knn (Gram matrix via v_wmma, 4 j-tiles/wave) -> gather edge feats (f16)
//   -> WMMA GEMM (+BN-stats atomics) -> BN finalize -> BN+leaky
//   -> WMMA GEMM (+stats) -> BN finalize -> fused BN+leaky+max-over-K
//   x4 layers, then proj GEMM, gate GEMM, softmax pool, tiny classifier.
// ---------------------------------------------------------------------------

#define NB    8
#define NPG   2048
#define NODES (NB * NPG)
#define KNN   20
#define MEDGE (NODES * KNN)
#define EMBED 1024

typedef __attribute__((ext_vector_type(16))) _Float16 v16h;
typedef __attribute__((ext_vector_type(8)))  _Float16 v8h;
typedef __attribute__((ext_vector_type(8)))  float    v8f;

__device__ __forceinline__ v16h cat8(v8h a, v8h b) {
    return __builtin_shufflevector(a, b, 0,1,2,3,4,5,6,7,8,9,10,11,12,13,14,15);
}

__device__ __forceinline__ v8f wmma_f16(v16h a, v16h b, v8f c) {
    return __builtin_amdgcn_wmma_f32_16x16x32_f16(false, a, false, b,
                                                  (short)0, c, false, false);
}

// ------------------------------ utility ------------------------------------

__global__ void zero_f32(float* p, int n) {
    int i = blockIdx.x * blockDim.x + threadIdx.x;
    if (i < n) p[i] = 0.f;
}

// f32 [node, C] -> f16 [node, ldx] zero-padded (ldx multiple of 32)
__global__ void to_half_pad(const float* __restrict__ x, _Float16* __restrict__ xh,
                            int C, int ldx) {
    int t = blockIdx.x * blockDim.x + threadIdx.x;
    if (t >= NODES * ldx) return;
    int node = t / ldx, c = t % ldx;
    xh[t] = (_Float16)((c < C) ? x[(size_t)node * C + c] : 0.f);
}

__global__ void sqnorm_kernel(const float* __restrict__ x, float* __restrict__ sq, int C) {
    int node = blockIdx.x * blockDim.x + threadIdx.x;
    if (node >= NODES) return;
    float s = 0.f;
    const float* p = x + (size_t)node * C;
    for (int c = 0; c < C; ++c) s += p[c] * p[c];
    sq[node] = s;
}

// W f32 [Kd, N] row-major -> Wt f16 [N, Kpad] (transposed, zero K-padded)
__global__ void convert_wt(const float* __restrict__ W, _Float16* __restrict__ Wt,
                           int Kd, int Kpad, int N) {
    int i = blockIdx.x * blockDim.x + threadIdx.x;
    if (i >= N * Kpad) return;
    int n = i / Kpad, kk = i % Kpad;
    Wt[i] = (_Float16)((kk < Kd) ? W[(size_t)kk * N + n] : 0.f);
}

// ------------------------------ knn ----------------------------------------
// One wave per 16-row i-tile of one graph. Gram tiles via WMMA, 4 j-tiles per
// outer step (A fragment reused 4x); top-20 via per-row LDS insertion sort.
__global__ __launch_bounds__(32)
void knn_kernel(const _Float16* __restrict__ xh, const float* __restrict__ sq,
                int* __restrict__ idxout, int ldx) {
    __shared__ float topd[16][KNN];
    __shared__ int   topi[16][KNN];
    __shared__ float dtile[16][64];
    const int lane = threadIdx.x;
    const int b  = blockIdx.x / (NPG / 16);
    const int it = blockIdx.x % (NPG / 16);
    const int i0 = it << 4;
    const _Float16* xb = xh + (size_t)b * NPG * ldx;
    if (lane < 16)
        for (int k = 0; k < KNN; ++k) { topd[lane][k] = 1e30f; topi[lane][k] = 0; }
    __syncthreads();

    const int rsel  = lane & 15;
    const int koff  = (lane & 16) ? 8 : 0;
    const int rbase = (lane & 16) ? 8 : 0;
    const _Float16* arow = xb + (size_t)(i0 + rsel) * ldx + koff;
    float sqi[8];
    for (int r = 0; r < 8; ++r) sqi[r] = sq[(size_t)b * NPG + i0 + rbase + r];

    for (int j0 = 0; j0 < NPG; j0 += 64) {
        const _Float16* brow[4];
#pragma unroll
        for (int t = 0; t < 4; ++t)
            brow[t] = xb + (size_t)(j0 + t * 16 + rsel) * ldx + koff;
        v8f acc[4] = {v8f{}, v8f{}, v8f{}, v8f{}};
        for (int ks = 0; ks < ldx; ks += 32) {
            v16h a = cat8(*(const v8h*)(arow + ks), *(const v8h*)(arow + ks + 16));
#pragma unroll
            for (int t = 0; t < 4; ++t) {
                v16h bb = cat8(*(const v8h*)(brow[t] + ks),
                               *(const v8h*)(brow[t] + ks + 16));
                acc[t] = wmma_f16(a, bb, acc[t]);
            }
        }
        const int jloc = lane & 15;
#pragma unroll
        for (int t = 0; t < 4; ++t) {
            const int jcol = t * 16 + jloc;
            const float sqj = sq[(size_t)b * NPG + j0 + jcol];
            for (int r = 0; r < 8; ++r) {
                int iloc = rbase + r;
                float d = sqi[r] + sqj - 2.f * acc[t][r];
                if (i0 + iloc == j0 + jcol) d = 1e30f;   // exclude self-loop
                dtile[iloc][jcol] = d;
            }
        }
        __syncthreads();
        if (lane < 16) {
            for (int jj = 0; jj < 64; ++jj) {
                float d = dtile[lane][jj];
                if (d < topd[lane][KNN - 1]) {
                    int pos = KNN - 1;
                    while (pos > 0 && topd[lane][pos - 1] > d) {
                        topd[lane][pos] = topd[lane][pos - 1];
                        topi[lane][pos] = topi[lane][pos - 1];
                        --pos;
                    }
                    topd[lane][pos] = d; topi[lane][pos] = j0 + jj;
                }
            }
        }
        __syncthreads();
    }
    if (lane < 16)
        for (int k = 0; k < KNN; ++k)
            idxout[((size_t)b * NPG + i0 + lane) * KNN + k] = topi[lane][k];
}

// ------------------------------ edge gather --------------------------------
// Edge m = (node, k): ef[m] = [x_i , x_j - x_i] in f16, zero-padded to ldK.
__global__ void gather_edges(const float* __restrict__ x, const int* __restrict__ idx,
                             _Float16* __restrict__ ef, int C, int ldK) {
    int m = blockIdx.x * blockDim.x + threadIdx.x;
    if (m >= MEDGE) return;
    int node = m / KNN, k = m % KNN;
    int b = node / NPG;
    int j = idx[(size_t)node * KNN + k];          // within-graph index
    const float* xi = x + (size_t)node * C;
    const float* xj = x + ((size_t)b * NPG + j) * C;
    _Float16* e = ef + (size_t)m * ldK;
    for (int c = 0; c < C; ++c) {
        float a = xi[c];
        e[c]     = (_Float16)a;
        e[C + c] = (_Float16)(xj[c] - a);
    }
    for (int c = 2 * C; c < ldK; ++c) e[c] = (_Float16)0.f;
}

// ------------------------------ WMMA GEMM + BN stats -----------------------
// A f16 [M, Kpad], Wt f16 [N, Kpad]; one wave per 16x64 output tile
// (4 accumulators, A fragment reused 4x per K-step). Writes Y f16 [M, N]
// (pre-BN, bias added) and atomically accumulates per-channel sum/sumsq.
// Requires N % 64 == 0, Kpad % 32 == 0, M % 16 == 0.
__global__ __launch_bounds__(256)
void wmma_gemm_stats(const _Float16* __restrict__ A,
                     const _Float16* __restrict__ Wt,
                     const float* __restrict__ bias,
                     _Float16* __restrict__ Y,
                     float* __restrict__ sums,
                     int M, int N, int Kpad, int do_stats) {
    int wave = blockIdx.x * 8 + (threadIdx.x >> 5);
    int lane = threadIdx.x & 31;
    int ngroups = N >> 6;
    if (wave >= (M >> 4) * ngroups) return;        // uniform per wave
    int mt = wave / ngroups, ng = wave % ngroups;
    int koff = (lane & 16) ? 8 : 0;
    const _Float16* arow = A + (size_t)((mt << 4) + (lane & 15)) * Kpad + koff;
    const _Float16* brow[4];
#pragma unroll
    for (int t = 0; t < 4; ++t)
        brow[t] = Wt + (size_t)((ng << 6) + t * 16 + (lane & 15)) * Kpad + koff;
    v8f acc[4] = {v8f{}, v8f{}, v8f{}, v8f{}};
    for (int ks = 0; ks < Kpad; ks += 32) {
        __builtin_prefetch(arow + ks + 128, 0, 1);
        v16h a = cat8(*(const v8h*)(arow + ks), *(const v8h*)(arow + ks + 16));
#pragma unroll
        for (int t = 0; t < 4; ++t) {
            v16h b = cat8(*(const v8h*)(brow[t] + ks),
                          *(const v8h*)(brow[t] + ks + 16));
            acc[t] = wmma_f16(a, b, acc[t]);
        }
    }
    const int rb = (mt << 4) + ((lane & 16) ? 8 : 0);
#pragma unroll
    for (int t = 0; t < 4; ++t) {
        const int ch = (ng << 6) + t * 16 + (lane & 15);
        const float bv = bias[ch];
        float lsum = 0.f, lsq = 0.f;
        for (int r = 0; r < 8; ++r) {
            float v = acc[t][r] + bv;
            Y[(size_t)(rb + r) * N + ch] = (_Float16)v;
            lsum += v; lsq += v * v;
        }
        if (do_stats) {
            lsum += __shfl_xor(lsum, 16, 32);
            lsq  += __shfl_xor(lsq, 16, 32);
            if (lane < 16) {
                atomicAdd(&sums[ch], lsum);
                atomicAdd(&sums[N + ch], lsq);
            }
        }
    }
}

// ------------------------------ BN finalize / apply ------------------------
__global__ void bn_finalize(const float* __restrict__ sums, float invCount,
                            const float* __restrict__ g, const float* __restrict__ be,
                            float* __restrict__ ss, int N) {
    int c = blockIdx.x * blockDim.x + threadIdx.x;
    if (c >= N) return;
    float mean = sums[c] * invCount;
    float var  = sums[N + c] * invCount - mean * mean;
    float sc = g[c] * rsqrtf(var + 1e-5f);
    ss[c]     = sc;
    ss[N + c] = be[c] - mean * sc;
}

__global__ void apply_bn_leaky(_Float16* __restrict__ Y, const float* __restrict__ ss,
                               int M, int N) {
    size_t i = (size_t)blockIdx.x * blockDim.x + threadIdx.x;
    if (i >= (size_t)M * N) return;
    int c = (int)(i % N);
    float v = (float)Y[i] * ss[c] + ss[N + c];
    Y[i] = (_Float16)(v >= 0.f ? v : 0.01f * v);
}

// fused BN + leaky + max over K neighbors -> next-layer features + concat feats
__global__ void maxk_kernel(const _Float16* __restrict__ Y2, const float* __restrict__ ss,
                            float* __restrict__ xnext, float* __restrict__ feats,
                            _Float16* __restrict__ featsh, int Cout, int featoff) {
    int t = blockIdx.x * blockDim.x + threadIdx.x;
    if (t >= NODES * Cout) return;
    int node = t / Cout, c = t % Cout;
    float sc = ss[c], sh = ss[Cout + c], m = -1e30f;
    const _Float16* y = Y2 + (size_t)node * KNN * Cout + c;
    for (int k = 0; k < KNN; ++k) {
        float v = (float)y[(size_t)k * Cout] * sc + sh;
        v = v >= 0.f ? v : 0.01f * v;
        m = fmaxf(m, v);
    }
    xnext[(size_t)node * Cout + c] = m;
    feats [(size_t)node * 512 + featoff + c] = m;
    featsh[(size_t)node * 512 + featoff + c] = (_Float16)m;
}

// ------------------------------ attention tail -----------------------------
__global__ void gate2_kernel(const _Float16* __restrict__ g1, const float* __restrict__ W,
                             const float* __restrict__ b, float* __restrict__ gpre,
                             float* __restrict__ sums) {
    int node = blockIdx.x * blockDim.x + threadIdx.x;
    if (node >= NODES) return;
    const _Float16* row = g1 + (size_t)node * 512;
    float s = b[0];
    for (int k = 0; k < 512; ++k) s += (float)row[k] * W[k];
    gpre[node] = s;
    atomicAdd(&sums[0], s);
    atomicAdd(&sums[1], s * s);
}

__global__ __launch_bounds__(256)
void softmax_pool(const float* __restrict__ gpre, const float* __restrict__ ss,
                  const _Float16* __restrict__ h, float* __restrict__ pooled) {
    __shared__ float a[NPG];
    __shared__ float red[256];
    int b = blockIdx.x, t = threadIdx.x;
    float mx = -1e30f;
    for (int n = t; n < NPG; n += 256) {
        float v = gpre[b * NPG + n] * ss[0] + ss[1];
        v = v >= 0.f ? v : 0.01f * v;
        a[n] = v;
        mx = fmaxf(mx, v);
    }
    red[t] = mx; __syncthreads();
    for (int s2 = 128; s2 > 0; s2 >>= 1) { if (t < s2) red[t] = fmaxf(red[t], red[t + s2]); __syncthreads(); }
    mx = red[0]; __syncthreads();
    float se = 0.f;
    for (int n = t; n < NPG; n += 256) { float e = __expf(a[n] - mx); a[n] = e; se += e; }
    red[t] = se; __syncthreads();
    for (int s2 = 128; s2 > 0; s2 >>= 1) { if (t < s2) red[t] += red[t + s2]; __syncthreads(); }
    float inv = 1.f / red[0];
    __syncthreads();
    for (int c = t; c < EMBED; c += 256) {
        float s = 0.f;
        const _Float16* hb = h + (size_t)b * NPG * EMBED + c;
        for (int n = 0; n < NPG; ++n) s += a[n] * (float)hb[(size_t)n * EMBED];
        pooled[b * EMBED + c] = s * inv;
    }
}

__global__ __launch_bounds__(512)
void cls_kernel(const float* __restrict__ pooled,
                const float* W1, const float* b1, const float* g1, const float* be1,
                const float* W2, const float* b2, const float* g2, const float* be2,
                const float* W3, const float* b3, float* __restrict__ out) {
    __shared__ float z1[8][512];
    __shared__ float z2[8][256];
    int t = threadIdx.x;
    if (t < 512) {
        float v[8];
        for (int r = 0; r < 8; ++r) {
            float s = b1[t];
            const float* p = pooled + r * 1024;
            for (int k = 0; k < 1024; ++k) s += p[k] * W1[(size_t)k * 512 + t];
            v[r] = s;
        }
        float m = 0.f; for (int r = 0; r < 8; ++r) m += v[r]; m *= 0.125f;
        float var = 0.f; for (int r = 0; r < 8; ++r) { float d = v[r] - m; var += d * d; } var *= 0.125f;
        float sc = g1[t] * rsqrtf(var + 1e-5f);
        for (int r = 0; r < 8; ++r) {
            float z = sc * (v[r] - m) + be1[t];
            z1[r][t] = z >= 0.f ? z : 0.01f * z;
        }
    }
    __syncthreads();
    if (t < 256) {
        float v[8];
        for (int r = 0; r < 8; ++r) {
            float s = b2[t];
            for (int k = 0; k < 512; ++k) s += z1[r][k] * W2[(size_t)k * 256 + t];
            v[r] = s;
        }
        float m = 0.f; for (int r = 0; r < 8; ++r) m += v[r]; m *= 0.125f;
        float var = 0.f; for (int r = 0; r < 8; ++r) { float d = v[r] - m; var += d * d; } var *= 0.125f;
        float sc = g2[t] * rsqrtf(var + 1e-5f);
        for (int r = 0; r < 8; ++r) {
            float z = sc * (v[r] - m) + be2[t];
            z2[r][t] = z >= 0.f ? z : 0.01f * z;
        }
    }
    __syncthreads();
    if (t < 8) {
        float s = b3[0];
        for (int k = 0; k < 256; ++k) s += z2[t][k] * W3[k];
        out[t] = s;
    }
}

// ------------------------------ host ---------------------------------------

extern "C" void kernel_launch(void* const* d_in, const int* in_sizes, int n_in,
                              void* d_out, int out_size, void* d_ws, size_t ws_size,
                              hipStream_t stream) {
    (void)in_sizes; (void)n_in; (void)out_size; (void)ws_size;
    const float* x_in = (const float*)d_in[0];

    // workspace carve-out (256B aligned)
    char* w = (char*)d_ws; size_t off = 0;
    auto carve = [&](size_t bytes) -> char* {
        char* p = w + off; off = (off + bytes + 255) & ~(size_t)255; return p;
    };
    float*    x_cur  = (float*)   carve((size_t)NODES * 256 * 4);
    _Float16* xh     = (_Float16*)carve((size_t)NODES * 256 * 2);
    float*    sq     = (float*)   carve((size_t)NODES * 4);
    int*      idxbuf = (int*)     carve((size_t)NODES * KNN * 4);
    float*    feats  = (float*)   carve((size_t)NODES * 512 * 4);
    _Float16* featsh = (_Float16*)carve((size_t)NODES * 512 * 2);
    _Float16* ef     = (_Float16*)carve((size_t)MEDGE * 256 * 2);  // edge feats / y2 / gate y
    _Float16* ybuf   = (_Float16*)carve((size_t)MEDGE * 256 * 2);  // y1/h1 / proj y -> hproj
    _Float16* wt     = (_Float16*)carve((size_t)1024 * 512 * 2);
    float*    sums   = (float*)   carve(2 * 1024 * 4);
    float*    ss     = (float*)   carve(2 * 1024 * 4);
    float*    gpre   = (float*)   carve((size_t)NODES * 4);
    float*    pooled = (float*)   carve((size_t)NB * EMBED * 4);

    const int Cin[4]  = {3, 64, 64, 128};
    const int Cout[4] = {64, 64, 128, 256};
    const int Ldx[4]  = {32, 64, 64, 128};   // Cin padded to mult of 32
    const int Kp [4]  = {32, 128, 128, 256}; // 2*Cin padded to mult of 32
    const int Foff[4] = {0, 64, 128, 256};

    const float* xl = x_in;
    for (int l = 0; l < 4; ++l) {
        const int Ci = Cin[l], Co = Cout[l], ldx = Ldx[l], kp = Kp[l];
        const float* W1  = (const float*)d_in[3 + l * 8 + 0];
        const float* b1  = (const float*)d_in[3 + l * 8 + 1];
        const float* g1  = (const float*)d_in[3 + l * 8 + 2];
        const float* be1 = (const float*)d_in[3 + l * 8 + 3];
        const float* W2  = (const float*)d_in[3 + l * 8 + 4];
        const float* b2  = (const float*)d_in[3 + l * 8 + 5];
        const float* g2  = (const float*)d_in[3 + l * 8 + 6];
        const float* be2 = (const float*)d_in[3 + l * 8 + 7];

        to_half_pad<<<(NODES * ldx + 255) / 256, 256, 0, stream>>>(xl, xh, Ci, ldx);
        sqnorm_kernel<<<(NODES + 255) / 256, 256, 0, stream>>>(xl, sq, Ci);
        knn_kernel<<<NODES / 16, 32, 0, stream>>>(xh, sq, idxbuf, ldx);
        gather_edges<<<(MEDGE + 127) / 128, 128, 0, stream>>>(xl, idxbuf, ef, Ci, kp);

        convert_wt<<<(Co * kp + 255) / 256, 256, 0, stream>>>(W1, wt, 2 * Ci, kp, Co);
        zero_f32<<<(2 * Co + 255) / 256, 256, 0, stream>>>(sums, 2 * Co);
        {
            int tiles = (MEDGE / 16) * (Co / 64);
            wmma_gemm_stats<<<(tiles + 7) / 8, 256, 0, stream>>>(ef, wt, b1, ybuf, sums,
                                                                 MEDGE, Co, kp, 1);
        }
        bn_finalize<<<(Co + 255) / 256, 256, 0, stream>>>(sums, 1.f / MEDGE, g1, be1, ss, Co);
        apply_bn_leaky<<<(int)(((size_t)MEDGE * Co + 255) / 256), 256, 0, stream>>>(ybuf, ss, MEDGE, Co);

        convert_wt<<<(Co * Co + 255) / 256, 256, 0, stream>>>(W2, wt, Co, Co, Co);
        zero_f32<<<(2 * Co + 255) / 256, 256, 0, stream>>>(sums, 2 * Co);
        {
            int tiles = (MEDGE / 16) * (Co / 64);
            wmma_gemm_stats<<<(tiles + 7) / 8, 256, 0, stream>>>(ybuf, wt, b2, ef, sums,
                                                                 MEDGE, Co, Co, 1);
        }
        bn_finalize<<<(Co + 255) / 256, 256, 0, stream>>>(sums, 1.f / MEDGE, g2, be2, ss, Co);
        maxk_kernel<<<(NODES * Co + 255) / 256, 256, 0, stream>>>(ef, ss, x_cur, feats,
                                                                  featsh, Co, Foff[l]);
        xl = x_cur;
    }

    // projection 512 -> 1024
    {
        const float* Wp  = (const float*)d_in[35];
        const float* bp  = (const float*)d_in[36];
        const float* gp  = (const float*)d_in[37];
        const float* bep = (const float*)d_in[38];
        convert_wt<<<(1024 * 512 + 255) / 256, 256, 0, stream>>>(Wp, wt, 512, 512, 1024);
        zero_f32<<<(2 * 1024 + 255) / 256, 256, 0, stream>>>(sums, 2 * 1024);
        int tiles = (NODES / 16) * (1024 / 64);
        wmma_gemm_stats<<<(tiles + 7) / 8, 256, 0, stream>>>(featsh, wt, bp, ybuf, sums,
                                                             NODES, 1024, 512, 1);
        bn_finalize<<<(1024 + 255) / 256, 256, 0, stream>>>(sums, 1.f / NODES, gp, bep, ss, 1024);
        apply_bn_leaky<<<(int)(((size_t)NODES * 1024 + 255) / 256), 256, 0, stream>>>(ybuf, ss, NODES, 1024);
    }
    // gate MLP 1024 -> 512 -> 1, softmax pool
    {
        const float* Wg1  = (const float*)d_in[39];
        const float* bg1  = (const float*)d_in[40];
        const float* gg1  = (const float*)d_in[41];
        const float* beg1 = (const float*)d_in[42];
        const float* Wg2  = (const float*)d_in[43];
        const float* bg2  = (const float*)d_in[44];
        const float* gg2  = (const float*)d_in[45];
        const float* beg2 = (const float*)d_in[46];
        convert_wt<<<(512 * 1024 + 255) / 256, 256, 0, stream>>>(Wg1, wt, 1024, 1024, 512);
        zero_f32<<<(2 * 512 + 255) / 256, 256, 0, stream>>>(sums, 2 * 512);
        int tiles = (NODES / 16) * (512 / 64);
        wmma_gemm_stats<<<(tiles + 7) / 8, 256, 0, stream>>>(ybuf, wt, bg1, ef, sums,
                                                             NODES, 512, 1024, 1);
        bn_finalize<<<(512 + 255) / 256, 256, 0, stream>>>(sums, 1.f / NODES, gg1, beg1, ss, 512);
        apply_bn_leaky<<<(int)(((size_t)NODES * 512 + 255) / 256), 256, 0, stream>>>(ef, ss, NODES, 512);
        zero_f32<<<1, 32, 0, stream>>>(sums, 2);
        gate2_kernel<<<NODES / 256, 256, 0, stream>>>(ef, Wg2, bg2, gpre, sums);
        bn_finalize<<<1, 32, 0, stream>>>(sums, 1.f / NODES, gg2, beg2, ss, 1);
        softmax_pool<<<NB, 256, 0, stream>>>(gpre, ss, ybuf, pooled);
    }
    // classifier head
    cls_kernel<<<1, 512, 0, stream>>>(pooled,
                                      (const float*)d_in[47], (const float*)d_in[48],
                                      (const float*)d_in[49], (const float*)d_in[50],
                                      (const float*)d_in[51], (const float*)d_in[52],
                                      (const float*)d_in[53], (const float*)d_in[54],
                                      (const float*)d_in[55], (const float*)d_in[56],
                                      (float*)d_out);
}
